// WeightQuantLinear_6176162971934
// MI455X (gfx1250) — compile-verified, matrix operand order checked by reading, not run on hardware
//
#include <hip/hip_runtime.h>

typedef __bf16 bf16;
typedef __attribute__((ext_vector_type(16))) __bf16 v16bf;
typedef __attribute__((ext_vector_type(8)))  float  v8f;
typedef unsigned int u32x4 __attribute__((ext_vector_type(4)));
typedef int          i32x8 __attribute__((ext_vector_type(8)));
typedef int          i32x4 __attribute__((ext_vector_type(4)));

#define BM 256            // block tile M
#define BN 128            // block tile N
#define BK 64             // K staged per iteration
#define LDP 72            // padded LDS row stride (elements): 144B rows, 16B aligned

// Load a 32-byte fragment (16 bf16) from LDS as two b128 reads.
__device__ __forceinline__ v16bf ld_frag(const bf16* p) {
    union { uint4 q[2]; v16bf v; } u;
    u.q[0] = *reinterpret_cast<const uint4*>(p);
    u.q[1] = *reinterpret_cast<const uint4*>(p + 8);
    return u.v;
}

__global__ __launch_bounds__(256) void wq_gemm_bf16_wmma(
    const bf16* __restrict__ X,          // [M, K] bf16
    const signed char* __restrict__ W,   // [N, K] int8
    const bf16* __restrict__ S,          // [N] bf16 per-channel scale
    bf16* __restrict__ O,                // [M, N] bf16
    int M, int N, int K)
{
    __shared__ bf16 sA[BM * LDP];        // 36 KB activation tile (padded rows)
    __shared__ bf16 sB[BN * LDP];        // 18 KB dequantized weight tile

    const int tid  = threadIdx.x;
    const int lane = tid & 31;
    const int wave = tid >> 5;
    const int wm   = wave & 3;           // 4 waves along M -> 64 rows each
    const int wn   = wave >> 2;          // 2 waves along N -> 64 cols each
    const int half = lane >> 4;          // 0: K 0..15 , 1: K 16..31
    const int r    = lane & 15;          // row within a 16x16 tile

    const int m0 = blockIdx.y * BM;
    const int n0 = blockIdx.x * BN;

    v8f acc[4][4];
#pragma unroll
    for (int i = 0; i < 4; ++i)
#pragma unroll
        for (int j = 0; j < 4; ++j)
            acc[i][j] = (v8f){0.f,0.f,0.f,0.f,0.f,0.f,0.f,0.f};

    // LDS byte offset of sA (generic LDS pointer: low 32 bits = LDS offset)
    const unsigned ldsA = (unsigned)(unsigned long long)(&sA[0]);

    for (int k0 = 0; k0 < K; k0 += BK) {
        // Prefetch next weight K-tile while this one is consumed.
        if (k0 + BK < K) {
            const int prow = tid >> 1;
            const int poff = (tid & 1) * 32;
            __builtin_prefetch(&W[(size_t)(n0 + prow) * K + k0 + BK + poff], 0, 1);
        }

        // ---- Stage A: 256 x 64 bf16 tile of x via the Tensor Data Mover ----
#if __has_builtin(__builtin_amdgcn_tensor_load_to_lds)
        if (wave == 0) {
            const unsigned long long ga =
                (unsigned long long)(const void*)X +
                ((unsigned long long)m0 * (unsigned)K + (unsigned)k0) * 2ull;
            // D# group 0: count=1 | lds_addr | global_addr | type=2
            u32x4 g0;
            g0[0] = 1u;                                   // count=1, user mode
            g0[1] = ldsA;                                 // LDS byte address
            g0[2] = (unsigned)ga;                         // global_addr[31:0]
            g0[3] = (unsigned)((ga >> 32) & 0x1FFFFFFull) // global_addr[56:32]
                    | (2u << 30);                         // type=2 ("image")
            // D# group 1: data_size=2B, pad: every 32 DWORDs insert 4 DWORDs
            //   -> LDS row stride = 128B + 16B = 144B = LDP elements
            i32x8 g1;
            g1[0] = (1 << 16)            // data_size = 2 bytes
                  | (1 << 20)            // pad_enable
                  | (4 << 22)            // pad_interval: 32 DWORDs (one row)
                  | (3 << 25);           // pad_amount: 4 DWORDs (16B)
            g1[1] = (int)(((unsigned)K & 0xFFFFu) << 16);              // tensor_dim0 lo
            g1[2] = (int)((((unsigned)K >> 16) & 0xFFFFu)
                  | (((unsigned)M & 0xFFFFu) << 16));                  // dim0 hi | dim1 lo
            g1[3] = (int)((((unsigned)M >> 16) & 0xFFFFu)
                  | ((unsigned)BK << 16));                             // dim1 hi | tile_dim0
            g1[4] = BM;                                                // tile_dim1 | tile_dim2=0
            g1[5] = K;                                                 // tensor_dim0_stride lo
            g1[6] = 0;                                                 // stride hi | dim1_stride lo
            g1[7] = 0;
            const i32x4 z4 = {0, 0, 0, 0};
#if defined(__clang_major__) && (__clang_major__ >= 23)
            const i32x8 z8 = {0, 0, 0, 0, 0, 0, 0, 0};
            __builtin_amdgcn_tensor_load_to_lds(g0, g1, z4, z4, z8, 0);
#else
            __builtin_amdgcn_tensor_load_to_lds(g0, g1, z4, z4, 0);
#endif
            __builtin_amdgcn_s_wait_tensorcnt(0);
        }
#else
        // Fallback: manual staged copy (8 x b128 per thread)
#pragma unroll
        for (int i = 0; i < 8; ++i) {
            const int idx = i * 256 + tid;
            const int row = idx >> 3;
            const int col = (idx & 7) << 3;
            const uint4 v = *reinterpret_cast<const uint4*>(
                &X[(size_t)(m0 + row) * K + k0 + col]);
            *reinterpret_cast<uint4*>(&sA[row * LDP + col]) = v;
        }
#endif

        // ---- Stage B: 128 x 64 int8 tile of W, dequantized int8 -> bf16 ----
#pragma unroll
        for (int i = 0; i < 2; ++i) {
            const int idx = i * 256 + tid;       // 0..511 chunks of 16 int8
            const int row = idx >> 2;            // 0..127
            const int col = (idx & 3) << 4;      // 0,16,32,48
            union { int4 raw; signed char c[16]; } w;
            w.raw = *reinterpret_cast<const int4*>(
                &W[(size_t)(n0 + row) * K + k0 + col]);
            union { uint4 q[2]; bf16 h[16]; } o;
#pragma unroll
            for (int e = 0; e < 16; ++e)
                o.h[e] = (bf16)(float)w.c[e];    // exact for |v| <= 127
            *reinterpret_cast<uint4*>(&sB[row * LDP + col])     = o.q[0];
            *reinterpret_cast<uint4*>(&sB[row * LDP + col + 8]) = o.q[1];
        }

        __syncthreads();

        // ---- Compute: two K=32 WMMA steps over the staged K=64 ----
#pragma unroll
        for (int ks = 0; ks < 2; ++ks) {
            const int kk = ks * 32 + half * 16;

            v16bf afrag[4];
#pragma unroll
            for (int i = 0; i < 4; ++i)
                afrag[i] = ld_frag(&sA[(wm * 64 + i * 16 + r) * LDP + kk]);

            v16bf bfrag[4];
#pragma unroll
            for (int j = 0; j < 4; ++j)
                bfrag[j] = ld_frag(&sB[(wn * 64 + j * 16 + r) * LDP + kk]);

#pragma unroll
            for (int i = 0; i < 4; ++i)
#pragma unroll
                for (int j = 0; j < 4; ++j)
                    acc[i][j] = __builtin_amdgcn_wmma_f32_16x16x32_bf16(
                        false, afrag[i], false, bfrag[j],
                        (short)0, acc[i][j], false, false);
        }

        __syncthreads();
    }

    // ---- Epilogue: per-channel scale in f32, convert to bf16, store ----
    const int rowBase = m0 + wm * 64;
    const int colBase = n0 + wn * 64;
#pragma unroll
    for (int j = 0; j < 4; ++j) {
        const int n = colBase + j * 16 + r;
        const float sc = (float)S[n];
#pragma unroll
        for (int i = 0; i < 4; ++i) {
#pragma unroll
            for (int v = 0; v < 8; ++v) {
                const int m = rowBase + i * 16 + half * 8 + v;
                O[(size_t)m * N + n] = (bf16)(acc[i][j][v] * sc);
            }
        }
    }
}

extern "C" void kernel_launch(void* const* d_in, const int* in_sizes, int n_in,
                              void* d_out, int out_size, void* d_ws, size_t ws_size,
                              hipStream_t stream) {
    const bf16*        X = (const bf16*)d_in[0];
    const signed char* W = (const signed char*)d_in[1];
    const bf16*        S = (const bf16*)d_in[2];
    bf16*              O = (bf16*)d_out;

    const int B = 4, Sq = 2048, K = 4096, N = 11008;
    const int M = B * Sq;                 // 8192

    dim3 grid(N / BN, M / BM);            // 86 x 32
    dim3 block(256);                      // 8 wave32 waves
    wq_gemm_bf16_wmma<<<grid, block, 0, stream>>>(X, W, S, O, M, N, K);
}
